// MBWCTNet_57827439674136
// MI455X (gfx1250) — compile-verified, matrix operand order checked by reading, not run on hardware
//
#include <hip/hip_runtime.h>
#include <hip/hip_bf16.h>

typedef _Float16 half_t;
typedef __attribute__((ext_vector_type(16))) _Float16 v16h;
typedef __attribute__((ext_vector_type(8)))  _Float16 v8h;
typedef __attribute__((ext_vector_type(8)))  float    v8f;

#define EPSF 1e-5f
#define LSEG 1020
#define NBAT 16
#define DMODEL 64
#define NHEAD 8
#define HDIM 8
#define NTOK (NBAT * LSEG)   // 16320, multiple of 16

// ---------------------------------------------------------------- utilities
__global__ __launch_bounds__(256)
void convert_f16_kernel(const float* __restrict__ a, half_t* __restrict__ o, int n) {
    int i = blockIdx.x * 256 + threadIdx.x;
    if (i < n) o[i] = (half_t)a[i];
}

// ---------------------------------------------------------------- conv stack
__global__ __launch_bounds__(256)
void conv1_kernel(const float* __restrict__ x, const float* __restrict__ w,
                  const float* __restrict__ bias, const float* __restrict__ g,
                  const float* __restrict__ bt, float* __restrict__ out) {
    const int L = 4080;
    int idx = blockIdx.x * 256 + threadIdx.x;
    if (idx >= NBAT * 32 * L) return;
    int l = idx % L; int co = (idx / L) % 32; int b = idx / (L * 32);
    float acc = bias[co];
    #pragma unroll
    for (int t = 0; t < 7; ++t) {
        int ll = l + t - 3;
        if (ll < 0 || ll >= L) continue;
        for (int ci = 0; ci < 12; ++ci)
            acc += x[((size_t)(b * 12 + ci)) * L + ll] * w[(co * 12 + ci) * 7 + t];
    }
    float v = acc * (g[co] * rsqrtf(1.f + EPSF)) + bt[co];
    out[idx] = fmaxf(v, 0.f);
}

__global__ __launch_bounds__(256)
void conv2_kernel(const float* __restrict__ h1, const float* __restrict__ w,
                  const float* __restrict__ bias, const float* __restrict__ g,
                  const float* __restrict__ bt, float* __restrict__ out) {
    const int L = 4080;
    int idx = blockIdx.x * 256 + threadIdx.x;
    if (idx >= NBAT * 32 * L) return;
    int l = idx % L; int co = (idx / L) % 32; int b = idx / (L * 32);
    float acc = bias[co];
    #pragma unroll
    for (int t = 0; t < 5; ++t) {
        int ll = l + t - 2;
        if (ll < 0 || ll >= L) continue;
        for (int ci = 0; ci < 32; ++ci)
            acc += h1[((size_t)(b * 32 + ci)) * L + ll] * w[(co * 32 + ci) * 5 + t];
    }
    float v = acc * (g[co] * rsqrtf(1.f + EPSF)) + bt[co];
    out[idx] = fmaxf(v, 0.f);
}

// conv3 (32->64,k3) + BN + ReLU fused with 4:1 adaptive average pool -> (B,64,1020)
__global__ __launch_bounds__(256)
void conv3_pool_kernel(const float* __restrict__ h2, const float* __restrict__ w,
                       const float* __restrict__ bias, const float* __restrict__ g,
                       const float* __restrict__ bt, float* __restrict__ hp) {
    const int L = 4080;
    int idx = blockIdx.x * 256 + threadIdx.x;
    if (idx >= NBAT * 64 * LSEG) return;
    int s = idx % LSEG; int co = (idx / LSEG) % 64; int b = idx / (LSEG * 64);
    float scale = g[co] * rsqrtf(1.f + EPSF);
    float acc4 = 0.f;
    #pragma unroll
    for (int r = 0; r < 4; ++r) {
        int l = s * 4 + r;
        float acc = bias[co];
        #pragma unroll
        for (int t = 0; t < 3; ++t) {
            int ll = l + t - 1;
            if (ll < 0 || ll >= L) continue;
            for (int ci = 0; ci < 32; ++ci)
                acc += h2[((size_t)(b * 32 + ci)) * L + ll] * w[(co * 32 + ci) * 3 + t];
        }
        acc4 += fmaxf(acc * scale + bt[co], 0.f);
    }
    hp[idx] = acc4 * 0.25f;
}

// ---------------------------------------------------------------- ECA
__global__ __launch_bounds__(256)
void eca_mean_kernel(const float* __restrict__ hp, float* __restrict__ y) {
    int i = blockIdx.x * 256 + threadIdx.x;      // (b,c): 1024
    if (i >= NBAT * 64) return;
    const float* p = hp + (size_t)i * LSEG;
    float s = 0.f;
    for (int l = 0; l < LSEG; ++l) s += p[l];
    y[i] = s * (1.f / LSEG);
}

// channel conv(k=3,SAME) + sigmoid gate + transpose to (B,L,C), f32 + f16 copies
__global__ __launch_bounds__(256)
void eca_apply_kernel(const float* __restrict__ hp, const float* __restrict__ y,
                      const float* __restrict__ ew, float* __restrict__ xt,
                      half_t* __restrict__ xt16) {
    int idx = blockIdx.x * 256 + threadIdx.x;    // (b,l,c)
    if (idx >= NBAT * LSEG * 64) return;
    int c = idx & 63; int l = (idx >> 6) % LSEG; int b = idx / (64 * LSEG);
    float conv = y[b * 64 + c] * ew[1];
    if (c > 0)  conv += y[b * 64 + c - 1] * ew[0];
    if (c < 63) conv += y[b * 64 + c + 1] * ew[2];
    float sg = 1.f / (1.f + __expf(-conv));
    float v = hp[((size_t)(b * 64 + c)) * LSEG + l] * sg;
    xt[idx] = v;
    xt16[idx] = (half_t)v;
}

// ---------------------------------------------------------------- WMMA GEMM
// C[M,N] = act(A[M,K] @ W[N,K]^T + bias). K in {64,128}; M,N multiples of 16.
// mode bit0 = relu, bit1 = f16 output (else f32). One 16x16 tile per wave.
__global__ __launch_bounds__(128)
void gemm16_kernel(const half_t* __restrict__ A, const half_t* __restrict__ W,
                   const float* __restrict__ bias, float* __restrict__ Cf,
                   half_t* __restrict__ Ch, int M, int N, int K, int mode) {
    int wave = blockIdx.x * 4 + (threadIdx.x >> 5);
    int lane = threadIdx.x & 31;
    int hf = lane >> 4, lr = lane & 15;
    int tilesN = N >> 4;
    int tm = wave / tilesN, tn = wave - tm * tilesN;
    if (tm >= (M >> 4)) return;
    int m0 = tm << 4, n0 = tn << 4;
    v8f c = {};
    const half_t* arow = A + (size_t)(m0 + lr) * K + hf * 8;
    const half_t* wrow = W + (size_t)(n0 + lr) * K + hf * 8;
    for (int k0 = 0; k0 < K; k0 += 32) {
        v8h alo = *(const v8h*)(arow + k0);
        v8h ahi = *(const v8h*)(arow + k0 + 16);
        v8h blo = *(const v8h*)(wrow + k0);
        v8h bhi = *(const v8h*)(wrow + k0 + 16);
        v16h a  = __builtin_shufflevector(alo, ahi, 0,1,2,3,4,5,6,7,8,9,10,11,12,13,14,15);
        v16h bb = __builtin_shufflevector(blo, bhi, 0,1,2,3,4,5,6,7,8,9,10,11,12,13,14,15);
        c = __builtin_amdgcn_wmma_f32_16x16x32_f16(false, a, false, bb, (short)0, c, false, false);
    }
    int n = n0 + lr;
    float bi = bias ? bias[n] : 0.f;
    #pragma unroll
    for (int r = 0; r < 8; ++r) {
        int m = m0 + r + hf * 8;
        float v = c[r] + bi;
        if (mode & 1) v = fmaxf(v, 0.f);
        if (mode & 2) Ch[(size_t)m * N + n] = (half_t)v;
        else          Cf[(size_t)m * N + n] = v;
    }
}

// ---------------------------------------------------------------- global attn
// Flash attention, one wave per (b, head, 16-query tile). qkv: (B,L,192) f16,
// q at +h*8, k at +64+h*8, v at +128+h*8. out: (B,L,64) f16.
__global__ __launch_bounds__(128)
void flash_attn_kernel(const half_t* __restrict__ qkv, half_t* __restrict__ out) {
    __shared__ _Float16 plds[4][16][16];
    int wid  = threadIdx.x >> 5;
    int wave = blockIdx.x * 4 + wid;             // exactly 16*8*64 = 8192 waves
    int qt = wave & 63; int h = (wave >> 6) & 7; int b = wave >> 9;
    int lane = threadIdx.x & 31;
    int hf = lane >> 4, lr = lane & 15;
    const float scale = 0.35355339059f;          // 1/sqrt(8)
    int q0 = qt * 16;

    // Q A-fragment (16 x 32, head_dim 8 valid, rest zero)
    v16h qa = {};
    int qrow = q0 + lr;
    if (qrow < LSEG) {
        const half_t* qp = qkv + ((size_t)(b * LSEG + qrow)) * 192 + h * 8;
        #pragma unroll
        for (int e = 0; e < 16; ++e) {
            int kk = (e & 7) + ((e >> 3) << 4) + hf * 8;
            if (kk < HDIM) qa[e] = qp[kk];
        }
    }

    float mrow[8], lrow[8];
    v8f o = {};
    #pragma unroll
    for (int r = 0; r < 8; ++r) { mrow[r] = -1e30f; lrow[r] = 0.f; }

    for (int kt = 0; kt < 64; ++kt) {
        // K B-fragment: column = key index within tile
        v16h kb = {};
        int key = kt * 16 + lr;
        if (key < LSEG) {
            const half_t* kp = qkv + ((size_t)(b * LSEG + key)) * 192 + 64 + h * 8;
            #pragma unroll
            for (int e = 0; e < 16; ++e) {
                int kk = (e & 7) + ((e >> 3) << 4) + hf * 8;
                if (kk < HDIM) kb[e] = kp[kk];
            }
        }
        v8f zero = {};
        v8f s = __builtin_amdgcn_wmma_f32_16x16x32_f16(false, qa, false, kb, (short)0, zero, false, false);

        bool colmask = (kt * 16 + lr) >= LSEG;
        #pragma unroll
        for (int r = 0; r < 8; ++r) {
            float v = s[r] * scale;
            s[r] = colmask ? -1e30f : v;
        }
        // online softmax per row (row = r + 8*hf; 16 key-columns across half-wave)
        #pragma unroll
        for (int r = 0; r < 8; ++r) {
            float v = s[r];
            v = fmaxf(v, __shfl_xor(v, 8, 16));
            v = fmaxf(v, __shfl_xor(v, 4, 16));
            v = fmaxf(v, __shfl_xor(v, 2, 16));
            v = fmaxf(v, __shfl_xor(v, 1, 16));
            float mnew = fmaxf(mrow[r], v);
            float sc = __expf(mrow[r] - mnew);
            float p  = __expf(s[r] - mnew);
            float ps = p;
            ps += __shfl_xor(ps, 8, 16);
            ps += __shfl_xor(ps, 4, 16);
            ps += __shfl_xor(ps, 2, 16);
            ps += __shfl_xor(ps, 1, 16);
            lrow[r] = lrow[r] * sc + ps;
            mrow[r] = mnew;
            o[r] *= sc;
            s[r] = p;
        }
        // transpose P (C-layout -> A-layout) through LDS
        __syncthreads();
        #pragma unroll
        for (int r = 0; r < 8; ++r) plds[wid][r + hf * 8][lr] = (half_t)s[r];
        __syncthreads();
        v16h pa = {};
        #pragma unroll
        for (int e = 0; e < 16; ++e) {
            int kk = (e & 7) + ((e >> 3) << 4) + hf * 8;
            if (kk < 16) pa[e] = plds[wid][lr][kk];
        }
        // V B-fragment: column = head-dim (8 valid), K = key within tile
        v16h vb = {};
        if (lr < HDIM) {
            #pragma unroll
            for (int e = 0; e < 16; ++e) {
                int kk = (e & 7) + ((e >> 3) << 4) + hf * 8;
                int key2 = kt * 16 + kk;
                if (kk < 16 && key2 < LSEG)
                    vb[e] = qkv[((size_t)(b * LSEG + key2)) * 192 + 128 + h * 8 + lr];
            }
        }
        o = __builtin_amdgcn_wmma_f32_16x16x32_f16(false, pa, false, vb, (short)0, o, false, false);
    }
    if (lr < HDIM) {
        #pragma unroll
        for (int r = 0; r < 8; ++r) {
            int m = q0 + r + hf * 8;
            if (m < LSEG) {
                float denom = lrow[r] > 0.f ? lrow[r] : 1.f;
                out[((size_t)(b * LSEG + m)) * DMODEL + h * 8 + lr] = (half_t)(o[r] / denom);
            }
        }
    }
}

// ---------------------------------------------------------------- local attn
// window=5, one thread per (b, window, head, query pos)
__global__ __launch_bounds__(256)
void local_attn_kernel(const half_t* __restrict__ qkv, half_t* __restrict__ out) {
    int idx = blockIdx.x * 256 + threadIdx.x;
    const int NW = LSEG / 5;                     // 204
    if (idx >= NBAT * NW * NHEAD * 5) return;
    int i = idx % 5; int h = (idx / 5) % NHEAD; int w = (idx / 40) % NW; int b = idx / (40 * NW);
    const half_t* base = qkv + ((size_t)(b * LSEG + w * 5)) * 192;
    float q[HDIM];
    #pragma unroll
    for (int d = 0; d < HDIM; ++d) q[d] = (float)base[(size_t)i * 192 + h * 8 + d];
    float s[5], mx = -1e30f;
    #pragma unroll
    for (int j = 0; j < 5; ++j) {
        float acc = 0.f;
        #pragma unroll
        for (int d = 0; d < HDIM; ++d) acc += q[d] * (float)base[(size_t)j * 192 + 64 + h * 8 + d];
        s[j] = acc * 0.35355339059f;
        mx = fmaxf(mx, s[j]);
    }
    float sum = 0.f;
    #pragma unroll
    for (int j = 0; j < 5; ++j) { s[j] = __expf(s[j] - mx); sum += s[j]; }
    float oacc[HDIM] = {0.f};
    #pragma unroll
    for (int j = 0; j < 5; ++j) {
        float p = s[j] / sum;
        #pragma unroll
        for (int d = 0; d < HDIM; ++d) oacc[d] += p * (float)base[(size_t)j * 192 + 128 + h * 8 + d];
    }
    #pragma unroll
    for (int d = 0; d < HDIM; ++d)
        out[((size_t)(b * LSEG + w * 5 + i)) * DMODEL + h * 8 + d] = (half_t)oacc[d];
}

// ---------------------------------------------------------------- LayerNorm
// y = LN(a + resid) over C=64, one wave per token; optional f16 copy
__global__ __launch_bounds__(256)
void ln_kernel(const float* __restrict__ a, const float* __restrict__ resid,
               const float* __restrict__ g, const float* __restrict__ bt,
               float* __restrict__ outf, half_t* __restrict__ outh, int ntok) {
    int wave = blockIdx.x * 8 + (threadIdx.x >> 5);
    int lane = threadIdx.x & 31;
    if (wave >= ntok) return;
    size_t base = (size_t)wave * 64;
    float x0 = a[base + lane]      + resid[base + lane];
    float x1 = a[base + lane + 32] + resid[base + lane + 32];
    float s = x0 + x1;
    s += __shfl_xor(s, 16, 32); s += __shfl_xor(s, 8, 32);
    s += __shfl_xor(s, 4, 32);  s += __shfl_xor(s, 2, 32); s += __shfl_xor(s, 1, 32);
    float mu = s * (1.f / 64.f);
    float d0 = x0 - mu, d1 = x1 - mu;
    float v = d0 * d0 + d1 * d1;
    v += __shfl_xor(v, 16, 32); v += __shfl_xor(v, 8, 32);
    v += __shfl_xor(v, 4, 32);  v += __shfl_xor(v, 2, 32); v += __shfl_xor(v, 1, 32);
    float inv = rsqrtf(v * (1.f / 64.f) + EPSF);
    float y0 = d0 * inv * g[lane]      + bt[lane];
    float y1 = d1 * inv * g[lane + 32] + bt[lane + 32];
    outf[base + lane] = y0; outf[base + lane + 32] = y1;
    if (outh) { outh[base + lane] = (half_t)y0; outh[base + lane + 32] = (half_t)y1; }
}

// fused = sigmoid(lw[c])*localn + (1-sigmoid)*globn ; f32 + f16 copies
__global__ __launch_bounds__(256)
void fuse_kernel(const float* __restrict__ localn, const float* __restrict__ globn,
                 const float* __restrict__ lw, float* __restrict__ f32,
                 half_t* __restrict__ f16, int n) {
    int idx = blockIdx.x * 256 + threadIdx.x;
    if (idx >= n) return;
    int c = idx & 63;
    float al = 1.f / (1.f + __expf(-lw[c]));
    float v = al * localn[idx] + (1.f - al) * globn[idx];
    f32[idx] = v; f16[idx] = (half_t)v;
}

// ---------------------------------------------------------------- head
__global__ __launch_bounds__(64)
void head_kernel(const float* __restrict__ outln, const float* __restrict__ dw,
                 const float* __restrict__ db, float* __restrict__ out) {
    __shared__ float feat[64];
    int b = blockIdx.x, c = threadIdx.x;
    float s = 0.f;
    for (int l = 0; l < LSEG; ++l) s += outln[((size_t)(b * LSEG + l)) * 64 + c];
    feat[c] = s * (1.f / LSEG);
    __syncthreads();
    if (c < 5) {
        float acc = db[c];
        for (int k = 0; k < 64; ++k) acc += feat[k] * dw[c * 64 + k];
        out[b * 5 + c] = acc;
    }
}

// ---------------------------------------------------------------- launcher
extern "C" void kernel_launch(void* const* d_in, const int* in_sizes, int n_in,
                              void* d_out, int out_size, void* d_ws, size_t ws_size,
                              hipStream_t stream) {
    const float* x       = (const float*)d_in[0];
    const float* c1w = (const float*)d_in[1];  const float* c1b = (const float*)d_in[2];
    const float* bn1g= (const float*)d_in[3];  const float* bn1b= (const float*)d_in[4];
    const float* c2w = (const float*)d_in[5];  const float* c2b = (const float*)d_in[6];
    const float* bn2g= (const float*)d_in[7];  const float* bn2b= (const float*)d_in[8];
    const float* c3w = (const float*)d_in[9];  const float* c3b = (const float*)d_in[10];
    const float* bn3g= (const float*)d_in[11]; const float* bn3b= (const float*)d_in[12];
    const float* ecaw= (const float*)d_in[13];
    const float* la_wqkv = (const float*)d_in[14]; const float* la_bqkv = (const float*)d_in[15];
    const float* la_wo   = (const float*)d_in[16]; const float* la_bo   = (const float*)d_in[17];
    const float* ga_wqkv = (const float*)d_in[18]; const float* ga_bqkv = (const float*)d_in[19];
    const float* ga_wo   = (const float*)d_in[20]; const float* ga_bo   = (const float*)d_in[21];
    const float* msat_lw = (const float*)d_in[22];
    const float* ffn_w1  = (const float*)d_in[23]; const float* ffn_b1 = (const float*)d_in[24];
    const float* ffn_w2  = (const float*)d_in[25]; const float* ffn_b2 = (const float*)d_in[26];
    const float* n1g = (const float*)d_in[27]; const float* n1b = (const float*)d_in[28];
    const float* n2g = (const float*)d_in[29]; const float* n2b = (const float*)d_in[30];
    const float* n3g = (const float*)d_in[31]; const float* n3b = (const float*)d_in[32];
    const float* dw  = (const float*)d_in[33]; const float* db  = (const float*)d_in[34];
    float* out = (float*)d_out;

    // bump allocator over d_ws (256B aligned)
    char* wsp = (char*)d_ws;
    auto alloc = [&](size_t bytes) -> void* {
        void* p = (void*)wsp;
        wsp += (bytes + 255) & ~(size_t)255;
        return p;
    };
    float*  h1    = (float*) alloc((size_t)NBAT*32*4080*4);
    float*  h2    = (float*) alloc((size_t)NBAT*32*4080*4);
    float*  hp    = (float*) alloc((size_t)NBAT*64*LSEG*4);
    float*  ymean = (float*) alloc((size_t)NBAT*64*4);
    float*  xt32  = (float*) alloc((size_t)NTOK*64*4);
    half_t* xt16  = (half_t*)alloc((size_t)NTOK*64*2);
    half_t* wqkvl16 = (half_t*)alloc(192*64*2);
    half_t* wqkvg16 = (half_t*)alloc(192*64*2);
    half_t* wol16   = (half_t*)alloc(64*64*2);
    half_t* wog16   = (half_t*)alloc(64*64*2);
    half_t* w1_16   = (half_t*)alloc(128*64*2);
    half_t* w2_16   = (half_t*)alloc(64*128*2);
    half_t* qkvl  = (half_t*)alloc((size_t)NTOK*192*2);
    half_t* qkvg  = (half_t*)alloc((size_t)NTOK*192*2);
    half_t* attl  = (half_t*)alloc((size_t)NTOK*64*2);
    half_t* attg  = (half_t*)alloc((size_t)NTOK*64*2);
    float*  pl    = (float*) alloc((size_t)NTOK*64*4);
    float*  pg    = (float*) alloc((size_t)NTOK*64*4);
    float*  localn= (float*) alloc((size_t)NTOK*64*4);
    float*  globn = (float*) alloc((size_t)NTOK*64*4);
    float*  fused32=(float*) alloc((size_t)NTOK*64*4);
    half_t* fused16=(half_t*)alloc((size_t)NTOK*64*2);
    half_t* ffn1  = (half_t*)alloc((size_t)NTOK*128*2);
    float*  ffn2  = (float*) alloc((size_t)NTOK*64*4);
    float*  outln = (float*) alloc((size_t)NTOK*64*4);

    auto blocks = [](int n) { return (n + 255) / 256; };

    // weight f16 conversion
    convert_f16_kernel<<<blocks(192*64), 256, 0, stream>>>(la_wqkv, wqkvl16, 192*64);
    convert_f16_kernel<<<blocks(192*64), 256, 0, stream>>>(ga_wqkv, wqkvg16, 192*64);
    convert_f16_kernel<<<blocks(64*64), 256, 0, stream>>>(la_wo, wol16, 64*64);
    convert_f16_kernel<<<blocks(64*64), 256, 0, stream>>>(ga_wo, wog16, 64*64);
    convert_f16_kernel<<<blocks(128*64), 256, 0, stream>>>(ffn_w1, w1_16, 128*64);
    convert_f16_kernel<<<blocks(64*128), 256, 0, stream>>>(ffn_w2, w2_16, 64*128);

    // conv stack + pool
    int nc = NBAT * 32 * 4080;
    conv1_kernel<<<blocks(nc), 256, 0, stream>>>(x, c1w, c1b, bn1g, bn1b, h1);
    conv2_kernel<<<blocks(nc), 256, 0, stream>>>(h1, c2w, c2b, bn2g, bn2b, h2);
    int np = NBAT * 64 * LSEG;
    conv3_pool_kernel<<<blocks(np), 256, 0, stream>>>(h2, c3w, c3b, bn3g, bn3b, hp);

    // ECA + transpose
    eca_mean_kernel<<<blocks(NBAT*64), 256, 0, stream>>>(hp, ymean);
    eca_apply_kernel<<<blocks(np), 256, 0, stream>>>(hp, ymean, ecaw, xt32, xt16);

    // QKV projections (WMMA): M=16320, N=192, K=64 -> f16
    int qkvTiles = (NTOK/16) * (192/16);
    gemm16_kernel<<<(qkvTiles+3)/4, 128, 0, stream>>>(xt16, wqkvl16, la_bqkv, nullptr, qkvl, NTOK, 192, 64, 2);
    gemm16_kernel<<<(qkvTiles+3)/4, 128, 0, stream>>>(xt16, wqkvg16, ga_bqkv, nullptr, qkvg, NTOK, 192, 64, 2);

    // attention
    local_attn_kernel<<<blocks(NBAT*(LSEG/5)*NHEAD*5), 256, 0, stream>>>(qkvl, attl);
    flash_attn_kernel<<<(NBAT*NHEAD*64)/4, 128, 0, stream>>>(qkvg, attg);

    // output projections (WMMA): N=64, K=64 -> f32
    int projTiles = (NTOK/16) * (64/16);
    gemm16_kernel<<<(projTiles+3)/4, 128, 0, stream>>>(attl, wol16, la_bo, pl, nullptr, NTOK, 64, 64, 0);
    gemm16_kernel<<<(projTiles+3)/4, 128, 0, stream>>>(attg, wog16, ga_bo, pg, nullptr, NTOK, 64, 64, 0);

    // LayerNorms (residual = xt)
    int lnBlocks = (NTOK + 7) / 8;
    ln_kernel<<<lnBlocks, 256, 0, stream>>>(pl, xt32, n1g, n1b, localn, nullptr, NTOK);
    ln_kernel<<<lnBlocks, 256, 0, stream>>>(pg, xt32, n2g, n2b, globn, nullptr, NTOK);

    // gated fusion
    fuse_kernel<<<blocks(NTOK*64), 256, 0, stream>>>(localn, globn, msat_lw, fused32, fused16, NTOK*64);

    // FFN (WMMA): 64 -> 128 (relu, f16) -> 64 (f32)
    int f1Tiles = (NTOK/16) * (128/16);
    gemm16_kernel<<<(f1Tiles+3)/4, 128, 0, stream>>>(fused16, w1_16, ffn_b1, nullptr, ffn1, NTOK, 128, 64, 3);
    gemm16_kernel<<<(projTiles+3)/4, 128, 0, stream>>>(ffn1, w2_16, ffn_b2, ffn2, nullptr, NTOK, 64, 128, 0);

    // final LN + head
    ln_kernel<<<lnBlocks, 256, 0, stream>>>(ffn2, fused32, n3g, n3b, outln, nullptr, NTOK);
    head_kernel<<<NBAT, 64, 0, stream>>>(outln, dw, db, out);

    (void)in_sizes; (void)n_in; (void)out_size; (void)ws_size;
}
